// LSTMBaseline_54365696032840
// MI455X (gfx1250) — compile-verified, matrix operand order checked by reading, not run on hardware
//
#include <hip/hip_runtime.h>
#include <hip/hip_bf16.h>

// Problem constants (match reference)
#define B_    128
#define SEQ_  512
#define DIN_  64
#define H_    1024
#define PRED_ 96
#define G4H_  4096   // 4*H

typedef __attribute__((ext_vector_type(16))) __bf16 v16bf;
typedef __attribute__((ext_vector_type(8)))  __bf16 v8bf;
typedef __attribute__((ext_vector_type(8)))  float  v8f;

// fp32 -> bf16 round-to-nearest-even
static __device__ __forceinline__ __bf16 f2bf(float f) {
  union { float f; unsigned u; } v; v.f = f;
  unsigned r = v.u + 0x7FFFu + ((v.u >> 16) & 1u);
  unsigned short h = (unsigned short)(r >> 16);
  return __builtin_bit_cast(__bf16, h);
}

// Load one 16x32 bf16 WMMA fragment (A layout; B is the same pattern with
// row = output column, since W is kept row-major (N x K)).
// Per ISA 7.12.2: lane&15 = matrix row, lane>>4 selects K octets
//   elements [0..7]  = K in [k0 + g*8,      k0 + g*8 + 7]
//   elements [8..15] = K in [k0 + 16 + g*8, k0 + 16 + g*8 + 7]
static __device__ __forceinline__ v16bf load_frag(const __bf16* __restrict__ base,
                                                  int pitch, int row, int k0, int lane) {
  int g = lane >> 4;
  const __bf16* p = base + (long)row * pitch + k0 + g * 8;
  v8bf lo = *(const v8bf*)(p);
  v8bf hi = *(const v8bf*)(p + 16);
  v16bf a;
#pragma unroll
  for (int i = 0; i < 8; ++i) { a[i] = lo[i]; a[8 + i] = hi[i]; }
  return a;
}

// One fused LSTM cell step:
//   gates(128x4096) = A1(128xK1) @ W1(4096xK1)^T + A2(128xK2) @ W2(4096xK2)^T + bias
//   i,f,g,o = split(gates); c' = sigm(f)*c + sigm(i)*tanh(g); h' = sigm(o)*tanh(c')
// Grid: (4, 16). Block: 256 threads = 8 waves = (2 M-subtiles) x (4 gates).
// Each block owns rows [bx*32, bx*32+32) and hidden cols [by*64, by*64+64)
// of ALL four gates -> fully local cell update via LDS staging.
__global__ __launch_bounds__(256)
void lstm_step_kernel(const __bf16* __restrict__ A1, int lda1,
                      const __bf16* __restrict__ W1, int K1,
                      const __bf16* __restrict__ A2, int lda2,
                      const __bf16* __restrict__ W2, int K2,
                      const float* __restrict__ bias,
                      const float* __restrict__ c_in,
                      float* __restrict__ c_out,
                      __bf16* __restrict__ h_out) {
  __shared__ float gsm[4][32][64];   // [gate][m][n] fp32 gate pre-activations

  const int tid  = threadIdx.x;
  const int lane = tid & 31;
  const int w    = tid >> 5;
  const int gate = w & 3;            // 0:i 1:f 2:g 3:o
  const int msub = w >> 2;           // 0/1
  const int ln15 = lane & 15;
  const int gl   = lane >> 4;

  const int arow  = blockIdx.x * 32 + msub * 16 + ln15;        // A row (batch)
  const int wrow0 = gate * H_ + blockIdx.y * 64;               // W row base (gate col)

  v8f acc[4] = {};                    // wave computes 16(M) x 64(N): 4 subtiles

  for (int k0 = 0; k0 < K1; k0 += 32) {
    v16bf a = load_frag(A1, lda1, arow, k0, lane);
#pragma unroll
    for (int j = 0; j < 4; ++j) {
      v16bf b = load_frag(W1, K1, wrow0 + j * 16 + ln15, k0, lane);
      acc[j] = __builtin_amdgcn_wmma_f32_16x16x32_bf16(false, a, false, b,
                                                       (short)0, acc[j], false, false);
    }
  }
  for (int k0 = 0; k0 < K2; k0 += 32) {
    v16bf a = load_frag(A2, lda2, arow, k0, lane);
#pragma unroll
    for (int j = 0; j < 4; ++j) {
      v16bf b = load_frag(W2, K2, wrow0 + j * 16 + ln15, k0, lane);
      acc[j] = __builtin_amdgcn_wmma_f32_16x16x32_bf16(false, a, false, b,
                                                       (short)0, acc[j], false, false);
    }
  }

  // Stage accumulators: C/D layout -> element (m = v + 8*gl, n = j*16 + ln15)
#pragma unroll
  for (int j = 0; j < 4; ++j)
#pragma unroll
    for (int v = 0; v < 8; ++v)
      gsm[gate][msub * 16 + v + 8 * gl][j * 16 + ln15] = acc[j][v];
  __syncthreads();

  // Pointwise cell update: 32*64 = 2048 elems, 8 per thread (uniform, no divergence)
  for (int e = tid; e < 32 * 64; e += 256) {
    int ml = e >> 6, nl = e & 63;
    int row = blockIdx.x * 32 + ml;         // batch index
    int col = blockIdx.y * 64 + nl;         // hidden index
    float iv = gsm[0][ml][nl] + bias[0 * H_ + col];
    float fv = gsm[1][ml][nl] + bias[1 * H_ + col];
    float gv = gsm[2][ml][nl] + bias[2 * H_ + col];
    float ov = gsm[3][ml][nl] + bias[3 * H_ + col];
    float is = 1.f / (1.f + __expf(-iv));
    float fs = 1.f / (1.f + __expf(-fv));
    float gt = tanhf(gv);
    float os = 1.f / (1.f + __expf(-ov));
    float c  = fs * c_in[row * H_ + col] + is * gt;
    c_out[row * H_ + col] = c;
    h_out[row * H_ + col] = f2bf(os * tanhf(c));
  }
}

// Projection: pred(128x64) = h1(128x1024) @ projW(64x1024)^T + proj_b
// Single block, 8 waves: wave w -> rows [16w,16w+16) x all 64 cols.
__global__ __launch_bounds__(256)
void proj_kernel(const __bf16* __restrict__ h1, const __bf16* __restrict__ Wp,
                 const float* __restrict__ pb, float* __restrict__ out, int step,
                 __bf16* __restrict__ pred_bf) {
  const int tid = threadIdx.x, lane = tid & 31, w = tid >> 5;
  const int ln15 = lane & 15, gl = lane >> 4;
  const int arow = w * 16 + ln15;
  v8f acc[4] = {};
  for (int k0 = 0; k0 < H_; k0 += 32) {
    v16bf a = load_frag(h1, H_, arow, k0, lane);
#pragma unroll
    for (int j = 0; j < 4; ++j) {
      v16bf b = load_frag(Wp, H_, j * 16 + ln15, k0, lane);
      acc[j] = __builtin_amdgcn_wmma_f32_16x16x32_bf16(false, a, false, b,
                                                       (short)0, acc[j], false, false);
    }
  }
#pragma unroll
  for (int j = 0; j < 4; ++j)
#pragma unroll
    for (int v = 0; v < 8; ++v) {
      int m = w * 16 + v + 8 * gl;
      int n = j * 16 + ln15;
      float val = acc[j][v] + pb[n];
      out[(long)m * (PRED_ * DIN_) + step * DIN_ + n] = val;   // (B, PRED, DIN)
      pred_bf[m * DIN_ + n] = f2bf(val);                       // next decoder input
    }
}

__global__ void f32_to_bf16_kernel(const float* __restrict__ src,
                                   __bf16* __restrict__ dst, int n) {
  int i = blockIdx.x * blockDim.x + threadIdx.x;
  if (i < n) dst[i] = f2bf(src[i]);
}

__global__ void zero_u32_kernel(unsigned* __restrict__ p, int n) {
  int i = blockIdx.x * blockDim.x + threadIdx.x;
  if (i < n) p[i] = 0u;
}

__global__ void init_decin_kernel(const float* __restrict__ x, __bf16* __restrict__ prd) {
  int i = blockIdx.x * blockDim.x + threadIdx.x;   // B*DIN
  if (i < B_ * DIN_) {
    int b = i >> 6, n = i & 63;
    prd[i] = f2bf(x[(long)b * SEQ_ * DIN_ + (SEQ_ - 1) * DIN_ + n]);
  }
}

__global__ void write_scalar_kernel(float* p) {
  if (blockIdx.x == 0 && threadIdx.x == 0) p[0] = 0.f;
}

extern "C" void kernel_launch(void* const* d_in, const int* in_sizes, int n_in,
                              void* d_out, int out_size, void* d_ws, size_t ws_size,
                              hipStream_t stream) {
  const float* x     = (const float*)d_in[0];
  const float* eWih0 = (const float*)d_in[1];
  const float* eWhh0 = (const float*)d_in[2];
  const float* eb0   = (const float*)d_in[3];
  const float* eWih1 = (const float*)d_in[4];
  const float* eWhh1 = (const float*)d_in[5];
  const float* eb1   = (const float*)d_in[6];
  const float* dWih0 = (const float*)d_in[7];
  const float* dWhh0 = (const float*)d_in[8];
  const float* db0   = (const float*)d_in[9];
  const float* dWih1 = (const float*)d_in[10];
  const float* dWhh1 = (const float*)d_in[11];
  const float* db1   = (const float*)d_in[12];
  const float* projW = (const float*)d_in[13];
  const float* projB = (const float*)d_in[14];
  float* out = (float*)d_out;

  // ---- workspace layout (256B aligned slabs) ----
  char* ws = (char*)d_ws;
  size_t off = 0;
  auto alloc = [&](size_t bytes) -> void* {
    void* p = ws + off;
    off = (off + bytes + 255) & ~(size_t)255;
    return p;
  };
  __bf16* xbf    = (__bf16*)alloc((size_t)B_ * SEQ_ * DIN_ * 2);
  __bf16* weWih0 = (__bf16*)alloc((size_t)G4H_ * DIN_ * 2);
  __bf16* weWhh0 = (__bf16*)alloc((size_t)G4H_ * H_ * 2);
  __bf16* weWih1 = (__bf16*)alloc((size_t)G4H_ * H_ * 2);
  __bf16* weWhh1 = (__bf16*)alloc((size_t)G4H_ * H_ * 2);
  __bf16* wdWih0 = (__bf16*)alloc((size_t)G4H_ * DIN_ * 2);
  __bf16* wdWhh0 = (__bf16*)alloc((size_t)G4H_ * H_ * 2);
  __bf16* wdWih1 = (__bf16*)alloc((size_t)G4H_ * H_ * 2);
  __bf16* wdWhh1 = (__bf16*)alloc((size_t)G4H_ * H_ * 2);
  __bf16* wProj  = (__bf16*)alloc((size_t)DIN_ * H_ * 2);
  // state region (zeroed each call) — keep contiguous
  size_t state_off = off;
  __bf16* h0[2]; __bf16* h1[2]; __bf16* prd[2];
  h0[0] = (__bf16*)alloc((size_t)B_ * H_ * 2);
  h0[1] = (__bf16*)alloc((size_t)B_ * H_ * 2);
  h1[0] = (__bf16*)alloc((size_t)B_ * H_ * 2);
  h1[1] = (__bf16*)alloc((size_t)B_ * H_ * 2);
  float* c0 = (float*)alloc((size_t)B_ * H_ * 4);
  float* c1 = (float*)alloc((size_t)B_ * H_ * 4);
  prd[0] = (__bf16*)alloc((size_t)B_ * DIN_ * 2);
  prd[1] = (__bf16*)alloc((size_t)B_ * DIN_ * 2);
  size_t state_bytes = off - state_off;

  // ---- one-time per-call prep (deterministic) ----
  auto conv = [&](const float* s, __bf16* d, int n) {
    f32_to_bf16_kernel<<<(n + 255) / 256, 256, 0, stream>>>(s, d, n);
  };
  conv(x,     xbf,    B_ * SEQ_ * DIN_);
  conv(eWih0, weWih0, G4H_ * DIN_);
  conv(eWhh0, weWhh0, G4H_ * H_);
  conv(eWih1, weWih1, G4H_ * H_);
  conv(eWhh1, weWhh1, G4H_ * H_);
  conv(dWih0, wdWih0, G4H_ * DIN_);
  conv(dWhh0, wdWhh0, G4H_ * H_);
  conv(dWih1, wdWih1, G4H_ * H_);
  conv(dWhh1, wdWhh1, G4H_ * H_);
  conv(projW, wProj,  DIN_ * H_);

  int zwords = (int)(state_bytes / 4);
  zero_u32_kernel<<<(zwords + 255) / 256, 256, 0, stream>>>((unsigned*)(ws + state_off), zwords);
  init_decin_kernel<<<(B_ * DIN_ + 255) / 256, 256, 0, stream>>>(x, prd[0]);
  write_scalar_kernel<<<1, 1, 0, stream>>>(out + (out_size - 1));  // trailing scalar 0

  const dim3 grid(4, 16);   // 128/32 x 1024/64

  // ---- encoder: 512 steps, layers 0+1 interleaved, h ping-pong ----
  int p = 0;
  for (int t = 0; t < SEQ_; ++t) {
    lstm_step_kernel<<<grid, 256, 0, stream>>>(
        xbf + (size_t)t * DIN_, SEQ_ * DIN_, weWih0, DIN_,
        h0[p], H_, weWhh0, H_,
        eb0, c0, c0, h0[1 - p]);
    lstm_step_kernel<<<grid, 256, 0, stream>>>(
        h0[1 - p], H_, weWih1, H_,
        h1[p], H_, weWhh1, H_,
        eb1, c1, c1, h1[1 - p]);
    p ^= 1;
  }
  // p is back to 0 here (512 flips)

  // ---- decoder: 96 steps, layers 0+1 + projection ----
  int pp = 0;   // prd[0] holds x[:, -1, :]
  for (int s = 0; s < PRED_; ++s) {
    lstm_step_kernel<<<grid, 256, 0, stream>>>(
        prd[pp], DIN_, wdWih0, DIN_,
        h0[p], H_, wdWhh0, H_,
        db0, c0, c0, h0[1 - p]);
    lstm_step_kernel<<<grid, 256, 0, stream>>>(
        h0[1 - p], H_, wdWih1, H_,
        h1[p], H_, wdWhh1, H_,
        db1, c1, c1, h1[1 - p]);
    proj_kernel<<<1, 256, 0, stream>>>(h1[1 - p], wProj, projB, out, s, prd[1 - pp]);
    p ^= 1;
    pp ^= 1;
  }

  (void)in_sizes; (void)n_in; (void)ws_size;
}
